// LightNetAttention_72344429134397
// MI455X (gfx1250) — matrix operand order repaired and checked by reading, not compiled, not run on hardware
//
#include <hip/hip_runtime.h>
#include <hip/hip_bf16.h>

typedef __bf16 bf16_t;
typedef __attribute__((ext_vector_type(16))) __bf16 v16bf;
typedef __attribute__((ext_vector_type(8)))  float  v8f;
typedef __attribute__((ext_vector_type(2)))  float  v2f;

constexpr int B_  = 4;
constexpr int N_  = 2048;
constexpr int D_  = 1024;
constexpr int H_  = 16;
constexpr int HD_ = 64;
constexpr int CHUNK_ = 64;
constexpr int NC_ = N_ / CHUNK_;          // 32 chunks
constexpr int BN_ = B_ * N_;              // 8192 rows
constexpr float LN_BASE = 9.210340371976184f; // ln(10000)

static __device__ __forceinline__ v8f wmma_bf16(v16bf a, v16bf b, v8f c) {
  return __builtin_amdgcn_wmma_f32_16x16x32_bf16(false, a, false, b, (short)0, c,
                                                 false, false);
}
static __device__ __forceinline__ v8f wmma_f32x4(v2f a, v2f b, v8f c) {
  return __builtin_amdgcn_wmma_f32_16x16x4_f32(false, a, false, b, (short)0, c,
                                               false, false);
}

static __device__ __forceinline__ float fast_rcp(float x) {
#if defined(__has_builtin) && __has_builtin(__builtin_amdgcn_rcpf)
  return __builtin_amdgcn_rcpf(x);
#else
  return 1.f / x;
#endif
}
static __device__ __forceinline__ float fast_sigmoid(float x) {
  return fast_rcp(1.f + __expf(-x));
}

// ---------------------------------------------------------------------------
// CDNA5 async global->LDS copy (ASYNCcnt tracked), with safe fallback.
// Builtin signature (from compiler): (v4i addrspace(1)*, v4i addrspace(3)*,
// imm offset, imm cpol).
// ---------------------------------------------------------------------------
#if defined(__has_builtin)
#if __has_builtin(__builtin_amdgcn_global_load_async_to_lds_b128)
#define HAVE_ASYNC_LDS 1
#endif
#endif

typedef int v4i_vs __attribute__((__vector_size__(4 * sizeof(int))));
typedef __attribute__((address_space(1))) v4i_vs glob_v4i;
typedef __attribute__((address_space(3))) v4i_vs lds_v4i;

static __device__ __forceinline__ void async_cp16(const bf16_t* g, bf16_t* l) {
#ifdef HAVE_ASYNC_LDS
  __builtin_amdgcn_global_load_async_to_lds_b128(
      (glob_v4i*)(void*)const_cast<bf16_t*>(g), (lds_v4i*)(void*)l, 0, 0);
#else
  *(int4*)l = *(const int4*)g;
#endif
}

#if defined(__has_builtin) && __has_builtin(__builtin_amdgcn_s_wait_asynccnt)
#define WAIT_ASYNC(n) __builtin_amdgcn_s_wait_asynccnt(n)
#else
#define WAIT_ASYNC(n) asm volatile("s_wait_asynccnt %0" ::"n"(n) : "memory")
#endif

// ---------------------------------------------------------------------------
// f32 -> bf16 conversion
// ---------------------------------------------------------------------------
__global__ void f2bf_kernel(const float* __restrict__ src,
                            bf16_t* __restrict__ dst, int n) {
  int i = blockIdx.x * blockDim.x + threadIdx.x;
  if (i < n) dst[i] = (bf16_t)src[i];
}

// ---------------------------------------------------------------------------
// Transposing f32 -> bf16 conversion: W[R][C] -> Wt[C][R]
// ---------------------------------------------------------------------------
__global__ __launch_bounds__(256) void f2bf_transpose_kernel(
    const float* __restrict__ W, bf16_t* __restrict__ Wt, int R, int C) {
  __shared__ float tile[32][33];
  int c0 = blockIdx.x * 32, r0 = blockIdx.y * 32;
  int tx = threadIdx.x & 31, ty = threadIdx.x >> 5;   // 32 x 8
  for (int i = ty; i < 32; i += 8)
    tile[i][tx] = W[(size_t)(r0 + i) * C + c0 + tx];
  __syncthreads();
  for (int i = ty; i < 32; i += 8)
    Wt[(size_t)(c0 + i) * R + r0 + tx] = (bf16_t)tile[tx][i];
}

// ---------------------------------------------------------------------------
// bf16 WMMA GEMM:  C[M x N] = A[M x K] @ B[K x N], with B supplied transposed
// (Bt[N][K], row-major).  Block tile 128x256, K-tile 64, 8 waves x (64x64).
// Double-buffered LDS filled by async global->LDS b128 copies.
// act: 0 = none, 1 = silu.  Exactly one of Cf / Cb is non-null.
// ---------------------------------------------------------------------------
#define TM 128
#define TN 256
#define TK 64
#define LSTR 72                                   // padded LDS row stride (bf16)
constexpr int A_OPS = (TM * TK * 2 / 16) / 256;   // 4 x 16B per thread
constexpr int B_OPS = (TN * TK * 2 / 16) / 256;   // 8 x 16B per thread
constexpr int TILE_OPS = A_OPS + B_OPS;           // 12
constexpr size_t GEMM_LDS = (size_t)2 * (TM + TN) * LSTR * sizeof(bf16_t); // 108KB

__global__ __launch_bounds__(256) void gemm_bf16_kernel(
    const bf16_t* __restrict__ A, const bf16_t* __restrict__ Bt,
    float* __restrict__ Cf, bf16_t* __restrict__ Cb,
    int M, int Nn, int K, int act) {
  extern __shared__ char gsm[];
  bf16_t* As = (bf16_t*)gsm;                    // [2][TM][LSTR]
  bf16_t* Bs = As + 2 * TM * LSTR;              // [2][TN][LSTR]

  const int tid  = threadIdx.x;
  const int lane = tid & 31;
  const int wave = tid >> 5;
  const int half = lane >> 4;
  const int r    = lane & 15;

  const int m0 = blockIdx.y * TM;
  const int n0 = blockIdx.x * TN;
  const int wm = (wave >> 2) * 64;              // 0 / 64
  const int wn = (wave & 3) * 64;               // 0..192

  v8f acc[4][4];
#pragma unroll
  for (int mt = 0; mt < 4; ++mt)
#pragma unroll
    for (int nt = 0; nt < 4; ++nt)
#pragma unroll
      for (int e = 0; e < 8; ++e) acc[mt][nt][e] = 0.f;

  auto issue_tile = [&](int k0, int buf) {
    bf16_t* Ad = As + buf * TM * LSTR;
    bf16_t* Bd = Bs + buf * TN * LSTR;
#pragma unroll
    for (int a = 0; a < A_OPS; ++a) {
      int c16 = tid + 256 * a;                  // 16B chunk id
      int row = c16 >> 3, col = (c16 & 7) * 8;
      async_cp16(A + (size_t)(m0 + row) * K + k0 + col,
                 Ad + row * LSTR + col);
    }
#pragma unroll
    for (int bop = 0; bop < B_OPS; ++bop) {
      int c16 = tid + 256 * bop;
      int row = c16 >> 3, col = (c16 & 7) * 8;
      async_cp16(Bt + (size_t)(n0 + row) * K + k0 + col,
                 Bd + row * LSTR + col);
    }
  };

  const int ntiles = K / TK;
  issue_tile(0, 0);
  for (int t = 0; t < ntiles; ++t) {
    if (t + 1 < ntiles) {
      issue_tile((t + 1) * TK, (t + 1) & 1);    // prefetch next tile
      WAIT_ASYNC(TILE_OPS);                     // tile t complete (in-order)
    } else {
      WAIT_ASYNC(0);
    }
    __syncthreads();

    const bf16_t* Ab = As + (t & 1) * TM * LSTR;
    const bf16_t* Bb = Bs + (t & 1) * TN * LSTR;
#pragma unroll
    for (int kk = 0; kk < 2; ++kk) {            // two WMMA K-steps per tile
      const int kb0 = kk * 32;
      v16bf af[4];
#pragma unroll
      for (int mt = 0; mt < 4; ++mt) {
        int row = wm + mt * 16 + r;
#pragma unroll
        for (int j = 0; j < 8; ++j) {
          int kb = kb0 + ((j >> 2) << 4) + half * 8 + ((j & 3) << 1);
          af[mt][2 * j]     = Ab[row * LSTR + kb];
          af[mt][2 * j + 1] = Ab[row * LSTR + kb + 1];
        }
      }
#pragma unroll
      for (int nt = 0; nt < 4; ++nt) {
        v16bf bfg;
        int col = wn + nt * 16 + r;
#pragma unroll
        for (int j = 0; j < 8; ++j) {
          int kb = kb0 + ((j >> 2) << 4) + half * 8 + ((j & 3) << 1);
          bfg[2 * j]     = Bb[col * LSTR + kb];
          bfg[2 * j + 1] = Bb[col * LSTR + kb + 1];
        }
#pragma unroll
        for (int mt = 0; mt < 4; ++mt)
          acc[mt][nt] = wmma_bf16(af[mt], bfg, acc[mt][nt]);
      }
    }
    __syncthreads();
  }

  // epilogue: C layout — VGPR i -> row (8*half + i), col = lane%16
#pragma unroll
  for (int mt = 0; mt < 4; ++mt)
#pragma unroll
    for (int nt = 0; nt < 4; ++nt)
#pragma unroll
      for (int i = 0; i < 8; ++i) {
        int gr = m0 + wm + mt * 16 + half * 8 + i;
        int gc = n0 + wn + nt * 16 + r;
        float v = acc[mt][nt][i];
        if (act == 1) v = v * fast_sigmoid(v);        // silu
        size_t idx = (size_t)gr * Nn + gc;
        if (Cf) Cf[idx] = v;
        else    Cb[idx] = (bf16_t)v;
      }
}

// ---------------------------------------------------------------------------
// Sequential cumlogsumexp scan along N for each (b, d) channel.
// ---------------------------------------------------------------------------
__global__ __launch_bounds__(256) void scan_kernel(
    const float* __restrict__ kraw, float* __restrict__ logfb,
    bf16_t* __restrict__ kd) {
  int ch = blockIdx.x * blockDim.x + threadIdx.x;   // 0 .. B*D-1
  int b = ch / D_;
  int d = ch % D_;
  const float* kr = kraw + (size_t)b * N_ * D_ + d;
  float m = 0.f;                                    // z0
  for (int t = 0; t < N_; ++t) {
    float x  = kr[(size_t)t * D_];
    float mx = fmaxf(m, x);
    float zn = mx + log1pf(__expf(fminf(m, x) - mx));
    size_t idx = ((size_t)(b * N_ + t)) * D_ + d;
    logfb[idx] = m - zn;
    kd[idx]    = (bf16_t)__expf(x - zn);
    m = zn;
  }
}

// ---------------------------------------------------------------------------
// u[bt][h] = x[bt] @ Wg1[:,h]
// ---------------------------------------------------------------------------
__global__ void gateu_kernel(const float* __restrict__ x,
                             const float* __restrict__ Wg1,
                             float* __restrict__ u) {
  int id = blockIdx.x * blockDim.x + threadIdx.x;
  if (id >= BN_ * H_) return;
  int bt = id / H_, hh = id % H_;
  float s = 0.f;
  for (int d = 0; d < D_; ++d) s += x[(size_t)bt * D_ + d] * Wg1[d * H_ + hh];
  u[id] = s;
}

// ---------------------------------------------------------------------------
// Chunked GLA.  One block per (b, h); 8 waves; recurrent state S (128x64 f32)
// kept in LDS across all 32 chunks.  All chunk matmuls use f32 WMMA 16x16x4.
// ---------------------------------------------------------------------------
constexpr int SQ_STRIDE = 132;
constexpr int SV_STRIDE = 68;
constexpr int LDS_FLOATS = 2 * 64 * SQ_STRIDE + 3 * 64 * SV_STRIDE
                         + 128 * SV_STRIDE;

__global__ __launch_bounds__(256) void gla_kernel(
    const bf16_t* __restrict__ qb, const bf16_t* __restrict__ kdb,
    const bf16_t* __restrict__ vb, const float* __restrict__ lf,
    float* __restrict__ obuf) {
  extern __shared__ float smem[];
  float* Sq  = smem;                      // [64][132]  q * e^G * lrpe
  float* Sk  = Sq  + 64 * SQ_STRIDE;      // [64][132]  k * e^-G * lrpe ; then ku
  float* Sv  = Sk  + 64 * SQ_STRIDE;      // [64][68]   v (f32)
  float* SG  = Sv  + 64 * SV_STRIDE;      // [64][68]   cumsum log_f in chunk
  float* SA  = SG  + 64 * SV_STRIDE;      // [64][68]   masked intra matrix
  float* SS  = SA  + 64 * SV_STRIDE;      // [128][68]  recurrent state

  const int tid  = threadIdx.x;
  const int lane = tid & 31;
  const int wave = tid >> 5;
  const int half = lane >> 4;
  const int r    = lane & 15;
  const int b = blockIdx.x / H_;
  const int h = blockIdx.x % H_;

  for (int i = tid; i < 128 * SV_STRIDE; i += 256) SS[i] = 0.f;
  __syncthreads();

  for (int cc = 0; cc < NC_; ++cc) {
    const int t0 = cc * CHUNK_;

#pragma unroll
    for (int e = 0; e < 16; ++e) {
      int li = tid + 256 * e;
      int c = li >> 6, j = li & 63;
      size_t gi = ((size_t)(b * N_ + t0 + c)) * D_ + h * HD_ + j;
      SG[c * SV_STRIDE + j] = lf[gi];
      Sv[c * SV_STRIDE + j] = (float)vb[gi];
    }
    __syncthreads();
    if (tid < 64) {
      float run = 0.f;
      for (int c = 0; c < 64; ++c) {
        run += SG[c * SV_STRIDE + tid];
        SG[c * SV_STRIDE + tid] = run;
      }
    }
    __syncthreads();
#pragma unroll
    for (int e = 0; e < 32; ++e) {
      int li = tid + 256 * e;
      int c = li >> 7, col = li & 127;
      int j = col & 63, issin = col >> 6;
      int pos = t0 + c;
      float inv = __expf(-(2.f * j / (float)HD_) * LN_BASE);
      float ang = (float)pos * inv;
      float tr  = issin ? __sinf(ang) : __cosf(ang);
      float G   = SG[c * SV_STRIDE + j];
      size_t gi = ((size_t)(b * N_ + pos)) * D_ + h * HD_ + j;
      Sq[c * SQ_STRIDE + col] = (float)qb[gi]  * tr * __expf(G);
      Sk[c * SQ_STRIDE + col] = (float)kdb[gi] * tr * __expf(-G);
    }
    __syncthreads();

    // o_inter = q_g @ S ; A = q_g @ k_g^T
    v8f acc[2], accA[2];
#pragma unroll
    for (int tt = 0; tt < 2; ++tt) {
#pragma unroll
      for (int e = 0; e < 8; ++e) { acc[tt][e] = 0.f; accA[tt][e] = 0.f; }
      int tile = wave * 2 + tt;
      int tm = tile >> 2, tn = tile & 3;
      for (int kk = 0; kk < 32; ++kk) {
        int kb = kk * 4 + 2 * half;
        v2f a, bS, bK;
        a.x  = Sq[(tm * 16 + r) * SQ_STRIDE + kb];
        a.y  = Sq[(tm * 16 + r) * SQ_STRIDE + kb + 1];
        bS.x = SS[kb * SV_STRIDE + tn * 16 + r];
        bS.y = SS[(kb + 1) * SV_STRIDE + tn * 16 + r];
        acc[tt] = wmma_f32x4(a, bS, acc[tt]);
        bK.x = Sk[(tn * 16 + r) * SQ_STRIDE + kb];
        bK.y = Sk[(tn * 16 + r) * SQ_STRIDE + kb + 1];
        accA[tt] = wmma_f32x4(a, bK, accA[tt]);
      }
#pragma unroll
      for (int i = 0; i < 8; ++i) {
        int cpos = tm * 16 + half * 8 + i;
        int spos = tn * 16 + r;
        SA[cpos * SV_STRIDE + spos] = (spos <= cpos) ? accA[tt][i] : 0.f;
      }
    }
    __syncthreads();

    // o += A @ v
#pragma unroll
    for (int tt = 0; tt < 2; ++tt) {
      int tile = wave * 2 + tt;
      int tm = tile >> 2, tn = tile & 3;
      for (int kk = 0; kk < 16; ++kk) {
        int kb = kk * 4 + 2 * half;
        v2f a, bV;
        a.x  = SA[(tm * 16 + r) * SV_STRIDE + kb];
        a.y  = SA[(tm * 16 + r) * SV_STRIDE + kb + 1];
        bV.x = Sv[kb * SV_STRIDE + tn * 16 + r];
        bV.y = Sv[(kb + 1) * SV_STRIDE + tn * 16 + r];
        acc[tt] = wmma_f32x4(a, bV, acc[tt]);
      }
#pragma unroll
      for (int i = 0; i < 8; ++i) {
        int c   = tm * 16 + half * 8 + i;
        int col = tn * 16 + r;
        obuf[((size_t)(b * N_ + t0 + c)) * D_ + h * HD_ + col] = acc[tt][i];
      }
    }
    __syncthreads();

    // rebuild Sk as ku = k_lrpe * exp(G_last - G)
#pragma unroll
    for (int e = 0; e < 32; ++e) {
      int li = tid + 256 * e;
      int c = li >> 7, col = li & 127;
      int j = col & 63, issin = col >> 6;
      int pos = t0 + c;
      float inv = __expf(-(2.f * j / (float)HD_) * LN_BASE);
      float ang = (float)pos * inv;
      float tr  = issin ? __sinf(ang) : __cosf(ang);
      float G   = SG[c * SV_STRIDE + j];
      float Gl  = SG[63 * SV_STRIDE + j];
      size_t gi = ((size_t)(b * N_ + pos)) * D_ + h * HD_ + j;
      Sk[c * SQ_STRIDE + col] = (float)kdb[gi] * tr * __expf(Gl - G);
    }
    __syncthreads();

    // S = exp(G_last)*S + ku^T @ v
#pragma unroll
    for (int tt = 0; tt < 4; ++tt) {
      int tile = wave * 4 + tt;
      int tm2 = tile >> 2, tn2 = tile & 3;
      v8f accS;
#pragma unroll
      for (int i = 0; i < 8; ++i) {
        int kdim = tm2 * 16 + half * 8 + i;
        int col  = tn2 * 16 + r;
        float Gl = SG[63 * SV_STRIDE + (kdim & 63)];
        accS[i] = SS[kdim * SV_STRIDE + col] * __expf(Gl);
      }
      for (int kk = 0; kk < 16; ++kk) {
        int kb = kk * 4 + 2 * half;
        v2f a, bV;
        a.x  = Sk[kb * SQ_STRIDE + tm2 * 16 + r];
        a.y  = Sk[(kb + 1) * SQ_STRIDE + tm2 * 16 + r];
        bV.x = Sv[kb * SV_STRIDE + tn2 * 16 + r];
        bV.y = Sv[(kb + 1) * SV_STRIDE + tn2 * 16 + r];
        accS = wmma_f32x4(a, bV, accS);
      }
#pragma unroll
      for (int i = 0; i < 8; ++i) {
        int kdim = tm2 * 16 + half * 8 + i;
        int col  = tn2 * 16 + r;
        SS[kdim * SV_STRIDE + col] = accS[i];
      }
    }
    __syncthreads();
  }
}

// ---------------------------------------------------------------------------
// o = o * sigmoid(u @ Wg2); RMSNorm over D; write bf16.  One block per row.
// ---------------------------------------------------------------------------
__global__ __launch_bounds__(256) void gatenorm_kernel(
    const float* __restrict__ obuf, const float* __restrict__ u,
    const float* __restrict__ Wg2, const float* __restrict__ norm_w,
    bf16_t* __restrict__ ob) {
  __shared__ float us[H_];
  __shared__ float red[256];
  int bt  = blockIdx.x;
  int tid = threadIdx.x;
  if (tid < H_) us[tid] = u[bt * H_ + tid];
  __syncthreads();
  float og[4];
  float ss = 0.f;
#pragma unroll
  for (int e = 0; e < 4; ++e) {
    int c = tid + 256 * e;
    float gl = 0.f;
#pragma unroll
    for (int hh = 0; hh < H_; ++hh) gl += us[hh] * Wg2[hh * D_ + c];
    float v = obuf[(size_t)bt * D_ + c] * fast_sigmoid(gl);
    og[e] = v;
    ss += v * v;
  }
  red[tid] = ss;
  __syncthreads();
  for (int s = 128; s > 0; s >>= 1) {
    if (tid < s) red[tid] += red[tid + s];
    __syncthreads();
  }
  float rs = rsqrtf(red[0] / (float)D_ + 1e-6f);
#pragma unroll
  for (int e = 0; e < 4; ++e) {
    int c = tid + 256 * e;
    ob[(size_t)bt * D_ + c] = (bf16_t)(og[e] * rs * norm_w[c]);
  }
}

// ---------------------------------------------------------------------------
// Host-side orchestration
// ---------------------------------------------------------------------------
extern "C" void kernel_launch(void* const* d_in, const int* in_sizes, int n_in,
                              void* d_out, int out_size, void* d_ws,
                              size_t ws_size, hipStream_t stream) {
  const float* x      = (const float*)d_in[0];
  const float* Wq     = (const float*)d_in[1];
  const float* Wk     = (const float*)d_in[2];
  const float* Wv     = (const float*)d_in[3];
  const float* Wo     = (const float*)d_in[4];
  const float* Wg1    = (const float*)d_in[5];
  const float* Wg2    = (const float*)d_in[6];
  const float* norm_w = (const float*)d_in[7];
  float* out = (float*)d_out;

  char* ws = (char*)d_ws;
  size_t off = 0;
  auto carve = [&](size_t bytes) {
    void* p = ws + off;
    off += (bytes + 255) & ~(size_t)255;
    return p;
  };
  bf16_t* xb    = (bf16_t*)carve((size_t)BN_ * D_ * 2);
  bf16_t* Wqt   = (bf16_t*)carve((size_t)D_ * D_ * 2);   // transposed [N][K]
  bf16_t* Wkt   = (bf16_t*)carve((size_t)D_ * D_ * 2);
  bf16_t* Wvt   = (bf16_t*)carve((size_t)D_ * D_ * 2);
  bf16_t* Wot   = (bf16_t*)carve((size_t)D_ * D_ * 2);
  bf16_t* qbuf  = (bf16_t*)carve((size_t)BN_ * D_ * 2);  // silu(x@Wq)
  float*  kraw  = (float*)carve((size_t)BN_ * D_ * 4);   // x@Wk
  bf16_t* vbuf  = (bf16_t*)carve((size_t)BN_ * D_ * 2);  // x@Wv
  float*  logfb = (float*)carve((size_t)BN_ * D_ * 4);   // log decay
  bf16_t* kdb   = (bf16_t*)carve((size_t)BN_ * D_ * 2);  // exp(k-z)
  float*  obuf  = (float*)carve((size_t)BN_ * D_ * 4);   // GLA output
  bf16_t* ob    = (bf16_t*)carve((size_t)BN_ * D_ * 2);  // gated+normed
  float*  u     = (float*)carve((size_t)BN_ * H_ * 4);   // x@Wg1

  // 1) convert x; transpose-convert the four big weights
  {
    int n = BN_ * D_;
    f2bf_kernel<<<(n + 255) / 256, 256, 0, stream>>>(x, xb, n);
    dim3 tg(D_ / 32, D_ / 32);
    f2bf_transpose_kernel<<<tg, 256, 0, stream>>>(Wq, Wqt, D_, D_);
    f2bf_transpose_kernel<<<tg, 256, 0, stream>>>(Wk, Wkt, D_, D_);
    f2bf_transpose_kernel<<<tg, 256, 0, stream>>>(Wv, Wvt, D_, D_);
    f2bf_transpose_kernel<<<tg, 256, 0, stream>>>(Wo, Wot, D_, D_);
  }

  // 2) projections (bf16 WMMA GEMMs, async double-buffered LDS)
  dim3 gg(D_ / TN, BN_ / TM);   // (4, 64)
  gemm_bf16_kernel<<<gg, 256, GEMM_LDS, stream>>>(xb, Wqt, nullptr, qbuf,
                                                  BN_, D_, D_, /*silu*/ 1);
  gemm_bf16_kernel<<<gg, 256, GEMM_LDS, stream>>>(xb, Wkt, kraw, nullptr,
                                                  BN_, D_, D_, 0);
  gemm_bf16_kernel<<<gg, 256, GEMM_LDS, stream>>>(xb, Wvt, nullptr, vbuf,
                                                  BN_, D_, D_, 0);

  // 3) gate inner projection
  gateu_kernel<<<(BN_ * H_ + 255) / 256, 256, 0, stream>>>(x, Wg1, u);

  // 4) cumlogsumexp scan along sequence
  scan_kernel<<<(B_ * D_) / 256, 256, 0, stream>>>(kraw, logfb, kdb);

  // 5) chunked GLA (f32 WMMA), one block per (b,h)
  gla_kernel<<<B_ * H_, 256, LDS_FLOATS * sizeof(float), stream>>>(
      qbuf, kdb, vbuf, logfb, obuf);

  // 6) gate * sigmoid + RMSNorm -> bf16
  gatenorm_kernel<<<BN_, 256, 0, stream>>>(obuf, u, Wg2, norm_w, ob);

  // 7) output projection -> f32 d_out
  gemm_bf16_kernel<<<gg, 256, GEMM_LDS, stream>>>(ob, Wot, out, nullptr,
                                                  BN_, D_, D_, 0);
}
